// Chunker_67740224193014
// MI455X (gfx1250) — matrix operand order repaired and checked by reading, not compiled, not run on hardware
//
#include <hip/hip_runtime.h>

typedef __attribute__((ext_vector_type(16))) __bf16 v16bf;
typedef __attribute__((ext_vector_type(8)))  float  v8f;

#define XW   520                 // padded x row stride (elements), 518 used
#define SY   520                 // Y row stride in bf16 elements (512 + 8 pad)
#define ZOFF (2 * 7 * XW)        // offset of the 512-elem zero region in xs
#define WPROJ_ELEMS (512*512)
#define WEMB_ELEMS  (32*128)

// One-instruction pack of two f32 -> packed bf16x2 via V_PERM_B32.
// +0x8000 gives round-half-up (differs from RNE only on exact ties).
__device__ __forceinline__ unsigned int pack2bf(float lo, float hi) {
    unsigned int ul = __float_as_uint(lo) + 0x8000u;
    unsigned int uh = __float_as_uint(hi) + 0x8000u;
    return __builtin_amdgcn_perm(uh, ul, 0x07060302u);  // {uh[31:16], ul[31:16]}
}
__device__ __forceinline__ unsigned short bf16_rn(float f) {
    return (unsigned short)((__float_as_uint(f) + 0x8000u) >> 16);
}

// constexpr im2col offset: k = ic*49 + kh*7 + kw -> element offset into xs.
// k >= 98 (zero padding of K to 128) points at the zeroed region.
__device__ __forceinline__ constexpr int im2col_off(int k) {
    return (k < 98) ? ((k / 49) * 7 * XW + ((k % 49) / 7) * XW + ((k % 49) % 7))
                    : ZOFF;
}

// ---------------- prep: convert weights to bf16 in workspace ----------------
__global__ void prep_weights(const float* __restrict__ w_proj,
                             const float* __restrict__ w_embed,
                             unsigned short* __restrict__ wsP,   // [512][512] bf16
                             unsigned short* __restrict__ wsE) { // [32][128]  bf16
    int i = blockIdx.x * 256 + threadIdx.x;
    if (i < WPROJ_ELEMS) {
        wsP[i] = bf16_rn(w_proj[i]);                 // [outC][inC] row-major
    }
    int j = i - WPROJ_ELEMS;
    if (j >= 0 && j < WEMB_ELEMS) {
        int c = j >> 7, k = j & 127;                 // K = ic*49 + kh*7 + kw, padded
        float v = 0.0f;
        if (k < 98) {
            int ic = k / 49, r = k % 49, kh = r / 7, kw = r % 7;
            v = w_embed[((c * 2 + ic) * 7 + kh) * 7 + kw];
        }
        wsE[j] = bf16_rn(v);
    }
}

// ---------------- fused conv7x7 + width_to_depth + 1x1 proj ----------------
__global__ __launch_bounds__(256)
void fused_kernel(const float* __restrict__ x,
                  const unsigned short* __restrict__ wP,   // bf16 [512][512]
                  const unsigned short* __restrict__ wE,   // bf16 [32][128]
                  const float* __restrict__ bE,
                  const float* __restrict__ bP,
                  float* __restrict__ out) {
    __shared__ unsigned short xs[ZOFF + 512];   // bf16 padded rows + zero region (15584 B)
    __shared__ unsigned short ys[32 * SY];      // Y tile, w2d layout             (33280 B)

    const int b = blockIdx.x >> 9;
    const int h = blockIdx.x & 511;
    const int tid   = threadIdx.x;
    const int wave  = tid >> 5;
    const int lane  = tid & 31;
    const int lmod  = lane & 15;
    const int lhalf = lane >> 4;

    // ---- stage padded x rows into LDS as bf16 (zero-pad h/w borders + zero region) ----
    const float* xb = x + (size_t)b * 2 * 512 * 512;
    for (int i = tid; i < ZOFF + 512; i += 256) {
        float v = 0.0f;
        if (i < ZOFF) {
            int ic  = i / (7 * XW);
            int rem = i % (7 * XW);
            int row = rem / XW;
            int col = rem % XW;
            int hr = h - 3 + row;
            int wc = col - 3;
            if (col < 518 && hr >= 0 && hr < 512 && wc >= 0 && wc < 512)
                v = xb[((size_t)ic * 512 + hr) * 512 + wc];
        }
        xs[i] = bf16_rn(v);
    }
    __syncthreads();

    // ================= stage 1: 7x7 conv via im2col WMMA =================
    // D[c, w] tile (mt, nt) -> Y[k' = (w%16)*32 + c][w' = nt]
    {
        // Hoist embed-weight A fragments (reused by all 4 n-tiles of this wave)
        union AF { v16bf v; uint4 q[2]; };
        AF aE[4][2];
        #pragma unroll
        for (int ks = 0; ks < 4; ++ks) {
            const int k0 = ks * 32 + lhalf * 8;
            #pragma unroll
            for (int mt = 0; mt < 2; ++mt) {
                const uint4* p = (const uint4*)(wE + (size_t)(mt * 16 + lmod) * 128 + k0);
                aE[ks][mt].q[0] = p[0];     // k0 .. k0+7
                aE[ks][mt].q[1] = p[2];     // k0+16 .. k0+23
            }
        }
        // Hoist embed bias
        float bEv[2][8];
        #pragma unroll
        for (int mt = 0; mt < 2; ++mt)
            #pragma unroll
            for (int r = 0; r < 8; ++r)
                bEv[mt][r] = bE[mt * 16 + lhalf * 8 + r];

        for (int i = 0; i < 4; ++i) {
            const int nt   = wave * 4 + i;          // 16-wide w tile
            const int ncol = nt * 16 + lmod;
            v8f acc[2];
            #pragma unroll
            for (int mt = 0; mt < 2; ++mt)
                acc[mt] = v8f{0.f, 0.f, 0.f, 0.f, 0.f, 0.f, 0.f, 0.f};

            #pragma unroll
            for (int ks = 0; ks < 4; ++ks) {
                // B fragment: gather bf16 im2col column from LDS.
                // Offsets constant-fold after unrolling; only lhalf is runtime.
                union { v16bf v; unsigned int u[8]; } bf;
                #pragma unroll
                for (int v2 = 0; v2 < 8; ++v2) {
                    const int t0 = 2 * v2, t1 = 2 * v2 + 1;
                    int o0 = lhalf ? im2col_off(ks * 32 + 8 + t0 + (t0 >= 8 ? 8 : 0))
                                   : im2col_off(ks * 32 + t0 + (t0 >= 8 ? 8 : 0));
                    int o1 = lhalf ? im2col_off(ks * 32 + 8 + t1 + (t1 >= 8 ? 8 : 0))
                                   : im2col_off(ks * 32 + t1 + (t1 >= 8 ? 8 : 0));
                    unsigned int e0 = xs[o0 + ncol];   // already bf16
                    unsigned int e1 = xs[o1 + ncol];
                    bf.u[v2] = e0 | (e1 << 16);        // v_lshl_or_b32
                }
                acc[0] = __builtin_amdgcn_wmma_f32_16x16x32_bf16(
                             false, aE[ks][0].v, false, bf.v, (short)0, acc[0], false, false);
                acc[1] = __builtin_amdgcn_wmma_f32_16x16x32_bf16(
                             false, aE[ks][1].v, false, bf.v, (short)0, acc[1], false, false);
            }

            // bias + pack + store into width_to_depth layout in LDS
            #pragma unroll
            for (int mt = 0; mt < 2; ++mt) {
                uint4 st;
                st.x = pack2bf(acc[mt][0] + bEv[mt][0], acc[mt][1] + bEv[mt][1]);
                st.y = pack2bf(acc[mt][2] + bEv[mt][2], acc[mt][3] + bEv[mt][3]);
                st.z = pack2bf(acc[mt][4] + bEv[mt][4], acc[mt][5] + bEv[mt][5]);
                st.w = pack2bf(acc[mt][6] + bEv[mt][6], acc[mt][7] + bEv[mt][7]);
                const int kp = lmod * 32 + mt * 16 + lhalf * 8;   // j*32 + c
                *(uint4*)(&ys[nt * SY + kp]) = st;                // 16B aligned
            }
        }
    }
    __syncthreads();

    // ================= stage 2: 1x1 projection GEMM (M=512,K=512,N=32) ====
    const int mbase = wave * 64;
    v8f acc[4][2];
    #pragma unroll
    for (int mt = 0; mt < 4; ++mt)
        #pragma unroll
        for (int nt = 0; nt < 2; ++nt)
            acc[mt][nt] = v8f{0.f, 0.f, 0.f, 0.f, 0.f, 0.f, 0.f, 0.f};

    for (int ks = 0; ks < 16; ++ks) {
        union { v16bf v; uint4 q[2]; } afr[4], bfr[2];
        const int k0 = ks * 32 + lhalf * 8;
        #pragma unroll
        for (int mt = 0; mt < 4; ++mt) {
            const uint4* p = (const uint4*)(wP + (size_t)(mbase + mt * 16 + lmod) * 512 + k0);
            afr[mt].q[0] = p[0];             // k0 .. k0+7
            afr[mt].q[1] = p[2];             // k0+16 .. k0+23
        }
        const int kb = ks * 32 + lhalf * 8;  // K kb..kb+7 and kb+16..kb+23 per lane
        #pragma unroll
        for (int nt = 0; nt < 2; ++nt) {
            const uint4* p = (const uint4*)(&ys[(nt * 16 + lmod) * SY + kb]);
            bfr[nt].q[0] = p[0];
            bfr[nt].q[1] = p[2];
        }
        #pragma unroll
        for (int mt = 0; mt < 4; ++mt)
            #pragma unroll
            for (int nt = 0; nt < 2; ++nt)
                acc[mt][nt] = __builtin_amdgcn_wmma_f32_16x16x32_bf16(
                    false, afr[mt].v, false, bfr[nt].v, (short)0,
                    acc[mt][nt], false, false);
    }

    // ---- epilogue: + b_proj, store z[b][outc][h][w'] ----
    #pragma unroll
    for (int mt = 0; mt < 4; ++mt) {
        const int ob = mbase + mt * 16 + lhalf * 8;
        #pragma unroll
        for (int nt = 0; nt < 2; ++nt) {
            const int wp_ = nt * 16 + lmod;
            #pragma unroll
            for (int r = 0; r < 8; ++r) {
                size_t idx = (((size_t)b * 512 + (ob + r)) * 512 + h) * 32 + wp_;
                out[idx] = acc[mt][nt][r] + bP[ob + r];
            }
        }
    }
}

extern "C" void kernel_launch(void* const* d_in, const int* in_sizes, int n_in,
                              void* d_out, int out_size, void* d_ws, size_t ws_size,
                              hipStream_t stream) {
    const float* x       = (const float*)d_in[0];
    const float* w_embed = (const float*)d_in[1];
    const float* b_embed = (const float*)d_in[2];
    const float* w_proj  = (const float*)d_in[3];
    const float* b_proj  = (const float*)d_in[4];
    float* out = (float*)d_out;

    unsigned short* wsP = (unsigned short*)d_ws;                 // 512 KB
    unsigned short* wsE = wsP + WPROJ_ELEMS;                     // +8 KB

    int prep_n = WPROJ_ELEMS + WEMB_ELEMS;
    prep_weights<<<(prep_n + 255) / 256, 256, 0, stream>>>(w_proj, w_embed, wsP, wsE);

    fused_kernel<<<8 * 512, 256, 0, stream>>>(x, wsP, wsE, b_embed, b_proj, out);
}
